// ActorCritic_44547400794436
// MI455X (gfx1250) — compile-verified, matrix operand order checked by reading, not run on hardware
//
#include <hip/hip_runtime.h>
#include <math.h>

typedef __attribute__((ext_vector_type(16))) _Float16 v16h;
typedef __attribute__((ext_vector_type(8)))  _Float16 v8h;
typedef __attribute__((ext_vector_type(8)))  float    v8f;

#define B_SZ     64
#define N_NODES  10000
#define N_TILES  625            // 10000 / 16
#define N_CAND   201
#define ROWS     (B_SZ * N_CAND)   // 12864
#define ROW_TILES (ROWS / 16)      // 804
#define HID      64
#define NODE_DIM 128
#define POOL_DIM 256
#define AC_IN    384
#define H_ACT    256

// ---------------- workspace layout (byte offsets) ----------------
#define WS_PSUM   0                      // 64*128 f32  = 32768
#define WS_PMAX   32768                  // 64*128 u32  = 32768
#define WS_POOL   65536                  // 64*256 f32  = 65536
#define WS_SCORE  131072                 // 12864 f32   = 51456
#define WS_AW1    182528                 // 384*256 f16 = 196608
#define WS_AW2    379136                 // 256*256 f16 = 131072
// total 510208 bytes

__device__ __forceinline__ v8f wmma16(v16h a, v16h b, v8f c) {
    return __builtin_amdgcn_wmma_f32_16x16x32_f16(false, a, false, b,
                                                  (short)0, c, false, false);
}

// K index inside a 32-wide K tile for (lane, element) per CDNA5 16-bit A/B layout
__device__ __forceinline__ int kk_of(int lane, int i) {
    int kb = (lane & 16) ? 8 : 0;
    return kb + ((i < 8) ? i : i + 8);   // i>=8 -> kb + 16 + (i-8)
}

// =====================================================================
// Kernel 0: swizzle actor weights into WMMA B-fragment f16 layout
// =====================================================================
__global__ void k0_swizzle(const float* __restrict__ a_w1,
                           const float* __restrict__ a_w2,
                           _Float16* __restrict__ aw1f,
                           _Float16* __restrict__ aw2f) {
    int idx = blockIdx.x * blockDim.x + threadIdx.x;   // 640 * 256 = 163840
    if (idx < 98304) {                                 // layer1: 12 kt x 16 nt
        int f = idx >> 9, lane = (idx >> 4) & 31, i = idx & 15;
        int kt = f / 16, nt = f % 16;
        int k = kt * 32 + kk_of(lane, i);
        int n = nt * 16 + (lane & 15);
        aw1f[idx] = (_Float16)a_w1[k * H_ACT + n];
    } else {
        int idx2 = idx - 98304;                        // layer2: 8 kt x 16 nt
        if (idx2 < 65536) {
            int f = idx2 >> 9, lane = (idx2 >> 4) & 31, i = idx2 & 15;
            int kt = f / 16, nt = f % 16;
            int k = kt * 32 + kk_of(lane, i);
            int n = nt * 16 + (lane & 15);
            aw2f[idx2] = (_Float16)a_w2[k * H_ACT + n];
        }
    }
}

// =====================================================================
// Kernel 1: feature extraction + pooling partials (WMMA)
// grid (16, 64), block 256 (8 waves), 5 tiles per wave
// =====================================================================
#define K1_TPW 5
__global__ __launch_bounds__(256)
void k1_features(const float* __restrict__ x,
                 const float* __restrict__ fe_w1, const float* __restrict__ fe_b1,
                 const float* __restrict__ fe_w2, const float* __restrict__ fe_b2,
                 float* __restrict__ psum, unsigned int* __restrict__ pmax) {
    __shared__ __attribute__((aligned(32))) float4   sW1[64];        // (w0,w1,w2,b1)
    __shared__ __attribute__((aligned(32))) _Float16 sBf[16 * 512];  // 16 frags x 32 lanes x 16

    const int tid = threadIdx.x;
    if (tid < 64)
        sW1[tid] = make_float4(fe_w1[tid], fe_w1[64 + tid], fe_w1[128 + tid], fe_b1[tid]);
    for (int idx = tid; idx < 16 * 512; idx += 256) {
        int f = idx >> 9, lane = (idx >> 4) & 31, i = idx & 15;
        int nt = f >> 1, kt = f & 1;
        int k = kt * 32 + kk_of(lane, i);
        int n = nt * 16 + (lane & 15);
        sBf[idx] = (_Float16)fe_w2[k * NODE_DIM + n];
    }
    __syncthreads();

    const int wave = tid >> 5, lane = tid & 31;
    const int m = lane & 15, kb = (lane & 16) ? 8 : 0;
    const int b = blockIdx.y;

    float acc_s[8][8], acc_m[8][8], biasv[8];
#pragma unroll
    for (int nt = 0; nt < 8; ++nt) {
        biasv[nt] = fe_b2[nt * 16 + (lane & 15)];
#pragma unroll
        for (int r = 0; r < 8; ++r) { acc_s[nt][r] = 0.f; acc_m[nt][r] = 0.f; }
    }

    const int tile0 = (blockIdx.x * 8 + wave) * K1_TPW;
    for (int t = 0; t < K1_TPW; ++t) {
        const int tile = tile0 + t;
        if (tile >= N_TILES) break;                    // wave-uniform
        const int node = tile * 16 + m;
        const float* xp = x + ((size_t)b * N_NODES + node) * 3;
        const float x0 = xp[0], x1 = xp[1], x2 = xp[2];

        v16h a0, a1;
#pragma unroll
        for (int i = 0; i < 16; ++i) {
            const int kk = kb + ((i < 8) ? i : i + 8);
            float4 w = sW1[kk];
            float h = fmaf(x2, w.z, fmaf(x1, w.y, fmaf(x0, w.x, w.w)));
            a0[i] = (_Float16)fmaxf(h, 0.f);
            float4 w2 = sW1[32 + kk];
            h = fmaf(x2, w2.z, fmaf(x1, w2.y, fmaf(x0, w2.x, w2.w)));
            a1[i] = (_Float16)fmaxf(h, 0.f);
        }
#pragma unroll
        for (int nt = 0; nt < 8; ++nt) {
            v8f c;
#pragma unroll
            for (int r = 0; r < 8; ++r) c[r] = biasv[nt];
            const v16h b0 = *(const v16h*)&sBf[(nt * 2 + 0) * 512 + lane * 16];
            const v16h b1 = *(const v16h*)&sBf[(nt * 2 + 1) * 512 + lane * 16];
            c = wmma16(a0, b0, c);
            c = wmma16(a1, b1, c);
#pragma unroll
            for (int r = 0; r < 8; ++r) {
                float d = fmaxf(c[r], 0.f);
                acc_s[nt][r] += d;
                acc_m[nt][r] = fmaxf(acc_m[nt][r], d);
            }
        }
    }
    // reduce: elems over m-rows, then half-wave combine, then global atomics
#pragma unroll
    for (int nt = 0; nt < 8; ++nt) {
        float s = 0.f, mx = 0.f;
#pragma unroll
        for (int r = 0; r < 8; ++r) { s += acc_s[nt][r]; mx = fmaxf(mx, acc_m[nt][r]); }
        s += __shfl_xor(s, 16);
        mx = fmaxf(mx, __shfl_xor(mx, 16));
        if (lane < 16) {
            atomicAdd(&psum[b * NODE_DIM + nt * 16 + lane], s);
            atomicMax(&pmax[b * NODE_DIM + nt * 16 + lane], __float_as_uint(mx));
        }
    }
}

// =====================================================================
// Kernel 2: finalize pooling + critic MLP.  grid 64, block 256
// =====================================================================
__global__ __launch_bounds__(256)
void k2_critic(const float* __restrict__ psum, const unsigned int* __restrict__ pmax,
               float* __restrict__ hpooled,
               const float* __restrict__ c_w1, const float* __restrict__ c_b1,
               const float* __restrict__ c_w2, const float* __restrict__ c_b2,
               const float* __restrict__ c_w3, const float* __restrict__ c_b3,
               float* __restrict__ vout) {
    __shared__ float v0[256], v1[256], red[256];
    const int j = threadIdx.x, b = blockIdx.x;
    float pv = (j < 128) ? psum[b * 128 + j] * (1.0f / (float)N_NODES)
                         : __uint_as_float(pmax[b * 128 + (j - 128)]);
    hpooled[b * 256 + j] = pv;
    v0[j] = pv;
    __syncthreads();
    float a = c_b1[j];
    for (int k = 0; k < 256; ++k) a = fmaf(v0[k], c_w1[k * 256 + j], a);
    v1[j] = tanhf(a);
    __syncthreads();
    a = c_b2[j];
    for (int k = 0; k < 256; ++k) a = fmaf(v1[k], c_w2[k * 256 + j], a);
    red[j] = tanhf(a) * c_w3[j];
    __syncthreads();
    for (int s = 128; s > 0; s >>= 1) {
        if (j < s) red[j] += red[j + s];
        __syncthreads();
    }
    if (j == 0) vout[b] = red[0] + c_b3[0];
}

// =====================================================================
// Kernel 3: actor MLP (WMMA).  grid 402, block 64 (2 waves), 1 tile/wave
// =====================================================================
__global__ __launch_bounds__(64)
void k3_actor(const float* __restrict__ x,
              const int* __restrict__ candidate,
              const unsigned char* __restrict__ mask,
              const float* __restrict__ fe_w1, const float* __restrict__ fe_b1,
              const float* __restrict__ fe_w2, const float* __restrict__ fe_b2,
              const float* __restrict__ hpooled,
              const _Float16* __restrict__ aw1f, const _Float16* __restrict__ aw2f,
              const float* __restrict__ a_b1, const float* __restrict__ a_b2,
              const float* __restrict__ a_w3, const float* __restrict__ a_b3,
              float* __restrict__ scores) {
    __shared__ __attribute__((aligned(32))) float4   sW1[64];
    __shared__ __attribute__((aligned(32))) _Float16 sBf[16 * 512];   // fe_w2 frags
    __shared__ float sW3[256];
    __shared__ __attribute__((aligned(32))) _Float16 sStage[2][16 * AC_IN + 16 * H_ACT];

    const int tid = threadIdx.x;
    if (tid < 64)
        sW1[tid] = make_float4(fe_w1[tid], fe_w1[64 + tid], fe_w1[128 + tid], fe_b1[tid]);
    for (int idx = tid; idx < 16 * 512; idx += 64) {
        int f = idx >> 9, lane = (idx >> 4) & 31, i = idx & 15;
        int nt = f >> 1, kt = f & 1;
        sBf[idx] = (_Float16)fe_w2[(kt * 32 + kk_of(lane, i)) * NODE_DIM + nt * 16 + (lane & 15)];
    }
    for (int idx = tid; idx < 256; idx += 64) sW3[idx] = a_w3[idx];
    __syncthreads();

    const int wave = tid >> 5, lane = tid & 31;
    const int m = lane & 15, kb = (lane & 16) ? 8 : 0;
    const int tileIdx = blockIdx.x * 2 + wave;          // 402*2 = 804 exactly
    const int row = tileIdx * 16 + m;
    const int bb = row / N_CAND, cc = row % N_CAND;

    _Float16* stIn  = &sStage[wave][0];                 // [16][384]
    _Float16* stAct = &sStage[wave][16 * AC_IN];        // [16][256]

    // ---- candidate node features via WMMA ----
    const int nd = candidate[bb * N_CAND + cc];
    const float* xp = x + ((size_t)bb * N_NODES + nd) * 3;
    const float x0 = xp[0], x1 = xp[1], x2 = xp[2];
    v16h a0, a1;
#pragma unroll
    for (int i = 0; i < 16; ++i) {
        const int kk = kb + ((i < 8) ? i : i + 8);
        float4 w = sW1[kk];
        float h = fmaf(x2, w.z, fmaf(x1, w.y, fmaf(x0, w.x, w.w)));
        a0[i] = (_Float16)fmaxf(h, 0.f);
        float4 w2 = sW1[32 + kk];
        h = fmaf(x2, w2.z, fmaf(x1, w2.y, fmaf(x0, w2.x, w2.w)));
        a1[i] = (_Float16)fmaxf(h, 0.f);
    }
#pragma unroll
    for (int nt = 0; nt < 8; ++nt) {
        v8f c;
        const float bias = fe_b2[nt * 16 + (lane & 15)];
#pragma unroll
        for (int r = 0; r < 8; ++r) c[r] = bias;
        c = wmma16(a0, *(const v16h*)&sBf[(nt * 2 + 0) * 512 + lane * 16], c);
        c = wmma16(a1, *(const v16h*)&sBf[(nt * 2 + 1) * 512 + lane * 16], c);
#pragma unroll
        for (int r = 0; r < 8; ++r) {
            const int mm = (lane < 16) ? r : r + 8;
            stIn[mm * AC_IN + nt * 16 + (lane & 15)] = (_Float16)fmaxf(c[r], 0.f);
        }
    }
    // ---- pooled part of input row (cols 128..383) ----
    {
        const int jb = (lane < 16) ? 0 : 128;
        const float* hp = &hpooled[bb * 256 + jb];
        for (int j = 0; j < 128; ++j)
            stIn[m * AC_IN + 128 + jb + j] = (_Float16)hp[j];
    }
    __syncthreads();

    // ---- layer 1: 384 -> 256, tanh ----
#pragma unroll 4
    for (int nt = 0; nt < 16; ++nt) {
        v8f acc;
        const float bias = a_b1[nt * 16 + (lane & 15)];
#pragma unroll
        for (int r = 0; r < 8; ++r) acc[r] = bias;
#pragma unroll
        for (int kt = 0; kt < 12; ++kt) {
            const v8h lo = *(const v8h*)&stIn[m * AC_IN + kt * 32 + kb];
            const v8h hi = *(const v8h*)&stIn[m * AC_IN + kt * 32 + 16 + kb];
            const v16h A = __builtin_shufflevector(lo, hi, 0,1,2,3,4,5,6,7,8,9,10,11,12,13,14,15);
            const v16h Bf = *(const v16h*)&aw1f[(kt * 16 + nt) * 512 + lane * 16];
            acc = wmma16(A, Bf, acc);
        }
#pragma unroll
        for (int r = 0; r < 8; ++r) {
            const int mm = (lane < 16) ? r : r + 8;
            stAct[mm * H_ACT + nt * 16 + (lane & 15)] = (_Float16)tanhf(acc[r]);
        }
    }
    __syncthreads();

    // ---- layer 2: 256 -> 256, tanh (output back into stIn as [16][256]) ----
#pragma unroll 4
    for (int nt = 0; nt < 16; ++nt) {
        v8f acc;
        const float bias = a_b2[nt * 16 + (lane & 15)];
#pragma unroll
        for (int r = 0; r < 8; ++r) acc[r] = bias;
#pragma unroll
        for (int kt = 0; kt < 8; ++kt) {
            const v8h lo = *(const v8h*)&stAct[m * H_ACT + kt * 32 + kb];
            const v8h hi = *(const v8h*)&stAct[m * H_ACT + kt * 32 + 16 + kb];
            const v16h A = __builtin_shufflevector(lo, hi, 0,1,2,3,4,5,6,7,8,9,10,11,12,13,14,15);
            const v16h Bf = *(const v16h*)&aw2f[(kt * 16 + nt) * 512 + lane * 16];
            acc = wmma16(A, Bf, acc);
        }
#pragma unroll
        for (int r = 0; r < 8; ++r) {
            const int mm = (lane < 16) ? r : r + 8;
            stIn[mm * H_ACT + nt * 16 + (lane & 15)] = (_Float16)tanhf(acc[r]);
        }
    }
    __syncthreads();

    // ---- layer 3: 256 -> 1 ----
    {
        const int kb2 = (lane < 16) ? 0 : 128;
        float s = 0.f;
        for (int j = 0; j < 128; ++j)
            s = fmaf((float)stIn[m * H_ACT + kb2 + j], sW3[kb2 + j], s);
        s += __shfl_xor(s, 16);
        if (lane < 16) {
            const int r2 = tileIdx * 16 + lane;
            const int b2 = r2 / N_CAND, c2 = r2 % N_CAND;
            const unsigned char mk = mask[b2 * N_CAND + c2];
            scores[r2] = mk ? (s + a_b3[0]) : -INFINITY;
        }
    }
}

// =====================================================================
// Kernel 4: masked softmax over candidates.  grid 64, block 256
// =====================================================================
__global__ __launch_bounds__(256)
void k4_softmax(const float* __restrict__ scores, float* __restrict__ pi) {
    __shared__ float red[256];
    const int tid = threadIdx.x, b = blockIdx.x;
    const float sc = (tid < N_CAND) ? scores[b * N_CAND + tid] : -INFINITY;
    red[tid] = sc;
    __syncthreads();
    for (int s = 128; s > 0; s >>= 1) {
        if (tid < s) red[tid] = fmaxf(red[tid], red[tid + s]);
        __syncthreads();
    }
    const float M = red[0];
    __syncthreads();
    const float e = (tid < N_CAND && sc > -INFINITY) ? expf(sc - M) : 0.f;
    red[tid] = e;
    __syncthreads();
    for (int s = 128; s > 0; s >>= 1) {
        if (tid < s) red[tid] += red[tid + s];
        __syncthreads();
    }
    const float inv = 1.0f / red[0];
    if (tid < N_CAND) pi[b * N_CAND + tid] = e * inv;
}

// =====================================================================
extern "C" void kernel_launch(void* const* d_in, const int* in_sizes, int n_in,
                              void* d_out, int out_size, void* d_ws, size_t ws_size,
                              hipStream_t stream) {
    const float* x         = (const float*)d_in[0];
    const int*   candidate = (const int*)d_in[1];
    const unsigned char* mask = (const unsigned char*)d_in[2];
    const float* fe_w1 = (const float*)d_in[3];
    const float* fe_b1 = (const float*)d_in[4];
    const float* fe_w2 = (const float*)d_in[5];
    const float* fe_b2 = (const float*)d_in[6];
    const float* a_w1  = (const float*)d_in[7];
    const float* a_b1  = (const float*)d_in[8];
    const float* a_w2  = (const float*)d_in[9];
    const float* a_b2  = (const float*)d_in[10];
    const float* a_w3  = (const float*)d_in[11];
    const float* a_b3  = (const float*)d_in[12];
    const float* c_w1  = (const float*)d_in[13];
    const float* c_b1  = (const float*)d_in[14];
    const float* c_w2  = (const float*)d_in[15];
    const float* c_b2  = (const float*)d_in[16];
    const float* c_w3  = (const float*)d_in[17];
    const float* c_b3  = (const float*)d_in[18];

    char* ws = (char*)d_ws;
    float*        psum    = (float*)(ws + WS_PSUM);
    unsigned int* pmax    = (unsigned int*)(ws + WS_PMAX);
    float*        hpooled = (float*)(ws + WS_POOL);
    float*        scores  = (float*)(ws + WS_SCORE);
    _Float16*     aw1f    = (_Float16*)(ws + WS_AW1);
    _Float16*     aw2f    = (_Float16*)(ws + WS_AW2);

    float* pi_out = (float*)d_out;            // [64*201]
    float* v_out  = (float*)d_out + ROWS;     // [64]

    hipMemsetAsync(ws, 0, 65536, stream);     // psum + pmax

    k0_swizzle<<<640, 256, 0, stream>>>(a_w1, a_w2, aw1f, aw2f);
    k1_features<<<dim3(16, B_SZ), 256, 0, stream>>>(x, fe_w1, fe_b1, fe_w2, fe_b2,
                                                    psum, pmax);
    k2_critic<<<B_SZ, 256, 0, stream>>>(psum, pmax, hpooled,
                                        c_w1, c_b1, c_w2, c_b2, c_w3, c_b3, v_out);
    k3_actor<<<ROW_TILES / 2, 64, 0, stream>>>(x, candidate, mask,
                                               fe_w1, fe_b1, fe_w2, fe_b2,
                                               hpooled, aw1f, aw2f,
                                               a_b1, a_b2, a_w3, a_b3, scores);
    k4_softmax<<<B_SZ, 256, 0, stream>>>(scores, pi_out);
}
